// ProteinMPNN_32487132627127
// MI455X (gfx1250) — matrix operand order repaired and checked by reading, not compiled, not run on hardware
//
#include <hip/hip_runtime.h>
#include <hip/hip_bf16.h>

#define BB 4
#define LLEN 1024
#define KK 48
#define HH 128
#define H3 384
#define FF 512
#define SCALE_F 30.0f
#define EPS_LN 1e-5f

// packed f16 weight segment offsets in d_ws (in halves)
#define OFF_W1   0
#define OFF_W2   49152
#define OFF_W3   65536
#define OFF_W11  81920
#define OFF_W12  131072
#define OFF_W13  147456
#define PACK_TOTAL 163840   // halves (= 327,680 bytes)

typedef _Float16 v16h __attribute__((ext_vector_type(16)));
typedef _Float16 v8h  __attribute__((ext_vector_type(8)));
typedef float    v8f  __attribute__((ext_vector_type(8)));

__device__ __forceinline__ float gelu_exact(float x) {
    return 0.5f * x * (1.0f + erff(x * 0.7071067811865476f));
}

__device__ __forceinline__ void zero_acc(v8f acc[3]) {
#pragma unroll
    for (int mt = 0; mt < 3; ++mt)
#pragma unroll
        for (int j = 0; j < 8; ++j) acc[mt][j] = 0.0f;
}

// ---------------------------------------------------------------------------
// Pack the six message-MLP weight matrices (fp32 [kdim x 128]) into f16,
// pre-swizzled into the B-fragment order used here (contiguous k = 16h + j):
//   packed[(((kc*8 + ntile)*32 + lane)*16) + j] = W[kc*32 + (lane&16?16:0) + j][16*ntile + lane%16]
// so each wave's B fragment per K-chunk is two coalesced b128 loads.
// ---------------------------------------------------------------------------
__global__ __launch_bounds__(256)
void pack_weights_kernel(const float* __restrict__ W1, const float* __restrict__ W2,
                         const float* __restrict__ W3, const float* __restrict__ W11,
                         const float* __restrict__ W12, const float* __restrict__ W13,
                         _Float16* __restrict__ dst) {
    int p = blockIdx.x * 256 + threadIdx.x;
    if (p >= PACK_TOTAL) return;
    const float* src;
    int base;
    if      (p < OFF_W2)  { src = W1;  base = OFF_W1;  }
    else if (p < OFF_W3)  { src = W2;  base = OFF_W2;  }
    else if (p < OFF_W11) { src = W3;  base = OFF_W3;  }
    else if (p < OFF_W12) { src = W11; base = OFF_W11; }
    else if (p < OFF_W13) { src = W12; base = OFF_W12; }
    else                  { src = W13; base = OFF_W13; }
    int q    = p - base;
    int j    = q & 15;
    int lane = (q >> 4) & 31;
    int nt   = (q >> 9) & 7;
    int kc   = q >> 12;
    int k = kc * 32 + ((lane & 16) ? 16 : 0) + j;
    int n = 16 * nt + (lane & 15);
    dst[p] = (_Float16)src[k * HH + n];
}

// ---------------------------------------------------------------------------
// 48x16 GEMM strip per wave. A: f16 LDS [48 x kdim] row-major (stride strideA
// halves) following the 16-bit A 16x32 lane layout. Wp: packed f16 B fragments.
// ---------------------------------------------------------------------------
__device__ __forceinline__ void gemm48(const _Float16* __restrict__ As, int strideA,
                                       int kdim, const _Float16* __restrict__ Wp,
                                       int wave, int lane, v8f acc[3]) {
    const int nkc = kdim >> 5;
    const _Float16* wp = Wp + (((0 * 8 + wave) * 32 + lane) << 4);
    for (int kc = 0; kc < nkc; ++kc) {
        // next-chunk prefetch (global_prefetch_b8); packed chunk stride = 8*32*16 halves
        __builtin_prefetch((const void*)(wp + 4096), 0, 0);
        v8h blo = *(const v8h*)(wp);       // coalesced global_load_b128
        v8h bhi = *(const v8h*)(wp + 8);
        v16h bf;
#pragma unroll
        for (int j = 0; j < 8; ++j) { bf[j] = blo[j]; bf[j + 8] = bhi[j]; }
#pragma unroll
        for (int mt = 0; mt < 3; ++mt) {
            const _Float16* ap = As + (mt * 16 + (lane & 15)) * strideA
                                    + kc * 32 + ((lane & 16) ? 16 : 0);
            v8h lo = *(const v8h*)(ap);    // ds_load_b128
            v8h hh = *(const v8h*)(ap + 8);
            v16h af;
#pragma unroll
            for (int j = 0; j < 8; ++j) { af[j] = lo[j]; af[j + 8] = hh[j]; }
            acc[mt] = __builtin_amdgcn_wmma_f32_16x16x32_f16(
                false, af, false, bf, (short)0, acc[mt], false, false);
        }
        wp += 4096;
    }
}

// bias + exact GELU + store f16 activation tile [48x128] per the C/D layout.
__device__ __forceinline__ void store_act(const v8f acc[3], const float* __restrict__ bias,
                                          _Float16* __restrict__ dst, int wave, int lane) {
    const int col = 16 * wave + (lane & 15);
    const float bb = bias[col];
    const int moff = (lane & 16) ? 8 : 0;
#pragma unroll
    for (int mt = 0; mt < 3; ++mt)
#pragma unroll
        for (int r = 0; r < 8; ++r) {
            int row = mt * 16 + r + moff;
            dst[row * HH + col] = (_Float16)gelu_exact(acc[mt][r] + bb);
        }
}

// ---------------------------------------------------------------------------
// Kernel A: node message MLP + K-reduction + LN1 + FFN + LN2 + mask -> h_V'
// ---------------------------------------------------------------------------
__global__ __launch_bounds__(256)
void node_update_kernel(const float* __restrict__ hV, const float* __restrict__ hE,
                        const int* __restrict__ Eidx, const float* __restrict__ maskV,
                        const float* __restrict__ maskA,
                        const _Float16* __restrict__ Wpack,
                        const float* __restrict__ b1, const float* __restrict__ b2,
                        const float* __restrict__ b3,
                        const float* __restrict__ n1g, const float* __restrict__ n1b,
                        const float* __restrict__ n2g, const float* __restrict__ n2b,
                        const float* __restrict__ Win, const float* __restrict__ bin,
                        const float* __restrict__ Wout, const float* __restrict__ bout,
                        float* __restrict__ outV) {
    __shared__ __align__(16) _Float16 Abuf[KK * H3];   // 36 KB arena
    __shared__ __align__(16) _Float16 act0[KK * HH];
    __shared__ __align__(16) _Float16 act1[KK * HH];
    __shared__ float red1[8], red2[8], stats[2];

    const int bl   = blockIdx.x;
    const int bIdx = bl / LLEN;
    const int tid  = threadIdx.x;
    const int wave = tid >> 5;
    const int lane = tid & 31;
    const float* hv_l = hV + (size_t)bl * HH;
    const long  ebase = (long)bl * KK;

    for (int i = tid; i < KK * H3; i += 256) {
        int k = i / H3, c = i - k * H3;
        float v;
        if (c < HH) {
            v = hv_l[c];
        } else if (c < 2 * HH) {
            v = hE[(ebase + k) * HH + (c - HH)];
        } else {
            int j = Eidx[ebase + k];
            v = hV[((long)bIdx * LLEN + j) * HH + (c - 2 * HH)];
        }
        Abuf[i] = (_Float16)v;
    }
    __syncthreads();

    v8f acc[3];
    zero_acc(acc);
    gemm48(Abuf, H3, H3, Wpack + OFF_W1, wave, lane, acc);
    store_act(acc, b1, act0, wave, lane);
    __syncthreads();

    zero_acc(acc);
    gemm48(act0, HH, HH, Wpack + OFF_W2, wave, lane, acc);
    store_act(acc, b2, act1, wave, lane);
    __syncthreads();

    zero_acc(acc);
    gemm48(act1, HH, HH, Wpack + OFF_W3, wave, lane, acc);

    // Masked sum over K=48 neighbors for this wave's 16 columns.
    float* dh = (float*)Abuf;
    const int col  = 16 * wave + (lane & 15);
    const int moff = (lane & 16) ? 8 : 0;
    const float b3c = b3[col];
    float s = 0.0f;
#pragma unroll
    for (int mt = 0; mt < 3; ++mt)
#pragma unroll
        for (int r = 0; r < 8; ++r) {
            int row = mt * 16 + r + moff;
            s += maskA[ebase + row] * (acc[mt][r] + b3c);
        }
    s += __shfl_xor(s, 16, 32);
    if (!(lane & 16)) dh[col] = s * (1.0f / SCALE_F);
    __syncthreads();

    // LN1
    float* xbuf  = dh + 128;
    float* ffbuf = dh + 256;
    if (tid < HH) xbuf[tid] = hv_l[tid] + dh[tid];
    __syncthreads();
    {
        float v = (tid < HH) ? xbuf[tid] : 0.0f;
        float s1 = v, s2 = v * v;
#pragma unroll
        for (int off = 16; off >= 1; off >>= 1) {
            s1 += __shfl_xor(s1, off, 32);
            s2 += __shfl_xor(s2, off, 32);
        }
        if (lane == 0) { red1[wave] = s1; red2[wave] = s2; }
    }
    __syncthreads();
    if (tid == 0) {
        float a = 0.0f, c = 0.0f;
        for (int i = 0; i < 8; ++i) { a += red1[i]; c += red2[i]; }
        float mean = a / (float)HH;
        stats[0] = mean;
        stats[1] = rsqrtf(c / (float)HH - mean * mean + EPS_LN);
    }
    __syncthreads();
    if (tid < HH) xbuf[tid] = (xbuf[tid] - stats[0]) * stats[1] * n1g[tid] + n1b[tid];
    __syncthreads();

    // FFN 128 -> 512 -> 128 (fp32 VALU; ~3% of FLOPs, weights L2-resident)
    for (int o = tid; o < FF; o += 256) {
        float sa = bin[o];
        for (int i = 0; i < HH; ++i) sa += xbuf[i] * Win[i * FF + o];
        ffbuf[o] = gelu_exact(sa);
    }
    __syncthreads();
    if (tid < HH) {
        float sa = bout[tid];
        for (int i = 0; i < FF; ++i) sa += ffbuf[i] * Wout[i * HH + tid];
        dh[tid] = xbuf[tid] + sa;
    }
    __syncthreads();

    // LN2 + mask_V
    {
        float v = (tid < HH) ? dh[tid] : 0.0f;
        float s1 = v, s2 = v * v;
#pragma unroll
        for (int off = 16; off >= 1; off >>= 1) {
            s1 += __shfl_xor(s1, off, 32);
            s2 += __shfl_xor(s2, off, 32);
        }
        if (lane == 0) { red1[wave] = s1; red2[wave] = s2; }
    }
    __syncthreads();
    if (tid == 0) {
        float a = 0.0f, c = 0.0f;
        for (int i = 0; i < 8; ++i) { a += red1[i]; c += red2[i]; }
        float mean = a / (float)HH;
        stats[0] = mean;
        stats[1] = rsqrtf(c / (float)HH - mean * mean + EPS_LN);
    }
    __syncthreads();
    if (tid < HH) {
        float yn = (dh[tid] - stats[0]) * stats[1] * n2g[tid] + n2b[tid];
        outV[(size_t)bl * HH + tid] = maskV[bl] * yn;
    }
}

// ---------------------------------------------------------------------------
// Kernel B: edge message MLP (with updated h_V) + residual + row-wise LN3 -> h_E'
// ---------------------------------------------------------------------------
__global__ __launch_bounds__(256)
void edge_update_kernel(const float* __restrict__ hVn, const float* __restrict__ hE,
                        const int* __restrict__ Eidx,
                        const _Float16* __restrict__ Wpack,
                        const float* __restrict__ b1, const float* __restrict__ b2,
                        const float* __restrict__ b3,
                        const float* __restrict__ n3g, const float* __restrict__ n3b,
                        float* __restrict__ outE) {
    __shared__ __align__(16) _Float16 Abuf[KK * H3];
    __shared__ __align__(16) _Float16 act0[KK * HH];
    __shared__ __align__(16) _Float16 act1[KK * HH];

    const int bl   = blockIdx.x;
    const int bIdx = bl / LLEN;
    const int tid  = threadIdx.x;
    const int wave = tid >> 5;
    const int lane = tid & 31;
    const float* hv_l = hVn + (size_t)bl * HH;
    const long  ebase = (long)bl * KK;

    for (int i = tid; i < KK * H3; i += 256) {
        int k = i / H3, c = i - k * H3;
        float v;
        if (c < HH) {
            v = hv_l[c];
        } else if (c < 2 * HH) {
            v = hE[(ebase + k) * HH + (c - HH)];
        } else {
            int j = Eidx[ebase + k];
            v = hVn[((long)bIdx * LLEN + j) * HH + (c - 2 * HH)];
        }
        Abuf[i] = (_Float16)v;
    }
    __syncthreads();

    v8f acc[3];
    zero_acc(acc);
    gemm48(Abuf, H3, H3, Wpack + OFF_W11, wave, lane, acc);
    store_act(acc, b1, act0, wave, lane);
    __syncthreads();

    zero_acc(acc);
    gemm48(act0, HH, HH, Wpack + OFF_W12, wave, lane, acc);
    store_act(acc, b2, act1, wave, lane);
    __syncthreads();

    zero_acc(acc);
    gemm48(act1, HH, HH, Wpack + OFF_W13, wave, lane, acc);

    // residual add into f32 LDS tile (reuse Abuf arena)
    float* msg = (float*)Abuf;
    const int col  = 16 * wave + (lane & 15);
    const int moff = (lane & 16) ? 8 : 0;
    const float b3c = b3[col];
#pragma unroll
    for (int mt = 0; mt < 3; ++mt)
#pragma unroll
        for (int r = 0; r < 8; ++r) {
            int row = mt * 16 + r + moff;
            msg[row * HH + col] = hE[(ebase + row) * HH + col] + acc[mt][r] + b3c;
        }
    __syncthreads();

    // Row-wise LayerNorm over H=128; one wave per row.
    for (int row = wave; row < KK; row += 8) {
        const float* mr = msg + row * HH;
        float t0 = mr[lane], t1 = mr[lane + 32], t2 = mr[lane + 64], t3 = mr[lane + 96];
        float s1 = t0 + t1 + t2 + t3;
        float s2 = t0 * t0 + t1 * t1 + t2 * t2 + t3 * t3;
#pragma unroll
        for (int off = 16; off >= 1; off >>= 1) {
            s1 += __shfl_xor(s1, off, 32);
            s2 += __shfl_xor(s2, off, 32);
        }
        float mean = s1 / (float)HH;
        float rstd = rsqrtf(s2 / (float)HH - mean * mean + EPS_LN);
        size_t base = (size_t)(ebase + row) * HH;
        outE[base + lane]      = (t0 - mean) * rstd * n3g[lane]      + n3b[lane];
        outE[base + lane + 32] = (t1 - mean) * rstd * n3g[lane + 32] + n3b[lane + 32];
        outE[base + lane + 64] = (t2 - mean) * rstd * n3g[lane + 64] + n3b[lane + 64];
        outE[base + lane + 96] = (t3 - mean) * rstd * n3g[lane + 96] + n3b[lane + 96];
    }
}

extern "C" void kernel_launch(void* const* d_in, const int* in_sizes, int n_in,
                              void* d_out, int out_size, void* d_ws, size_t ws_size,
                              hipStream_t stream) {
    const float* hV    = (const float*)d_in[0];
    const float* hE    = (const float*)d_in[1];
    const int*   Eidx  = (const int*)  d_in[2];
    const float* maskV = (const float*)d_in[3];
    const float* maskA = (const float*)d_in[4];
    const float* W1  = (const float*)d_in[5];
    const float* b1  = (const float*)d_in[6];
    const float* W2  = (const float*)d_in[7];
    const float* b2  = (const float*)d_in[8];
    const float* W3  = (const float*)d_in[9];
    const float* b3  = (const float*)d_in[10];
    const float* W11 = (const float*)d_in[11];
    const float* b11 = (const float*)d_in[12];
    const float* W12 = (const float*)d_in[13];
    const float* b12 = (const float*)d_in[14];
    const float* W13 = (const float*)d_in[15];
    const float* b13 = (const float*)d_in[16];
    const float* n1g = (const float*)d_in[17];
    const float* n1b = (const float*)d_in[18];
    const float* n2g = (const float*)d_in[19];
    const float* n2b = (const float*)d_in[20];
    const float* n3g = (const float*)d_in[21];
    const float* n3b = (const float*)d_in[22];
    const float* Win  = (const float*)d_in[23];
    const float* bin  = (const float*)d_in[24];
    const float* Wout = (const float*)d_in[25];
    const float* bout = (const float*)d_in[26];

    _Float16* Wpack = (_Float16*)d_ws;                 // 327,680 B of scratch
    float* outV = (float*)d_out;                       // [B,L,H]
    float* outE = outV + (size_t)BB * LLEN * HH;       // [B,L,K,H]

    pack_weights_kernel<<<(PACK_TOTAL + 255) / 256, 256, 0, stream>>>(
        W1, W2, W3, W11, W12, W13, Wpack);

    dim3 grid(BB * LLEN), block(256);
    node_update_kernel<<<grid, block, 0, stream>>>(
        hV, hE, Eidx, maskV, maskA, Wpack, b1, b2, b3,
        n1g, n1b, n2g, n2b, Win, bin, Wout, bout, outV);
    edge_update_kernel<<<grid, block, 0, stream>>>(
        outV, hE, Eidx, Wpack, b11, b12, b13, n3g, n3b, outE);
}
